// MetricConvBlock_41120016892881
// MI455X (gfx1250) — matrix operand order repaired and checked by reading, not compiled, not run on hardware
//
#include <hip/hip_runtime.h>

// ---------------------------------------------------------------------------
// MetricConvBlock for MI455X (gfx1250, wave32).
// Pipeline: vertex normals -> per-edge metric weight -> bf16 WMMA feature GEMM
//           -> edge-weighted scatter (f32 atomics, L2-resident) -> column
//           normalization + ELU.
// Round 2: features pre-converted to bf16 so the WMMA inner loop is pure
//          b128-load + v_wmma (no per-kstep cvt VALU).
// ---------------------------------------------------------------------------

#define CIN  128
#define COUT 128
#define HDIM 32

typedef __attribute__((ext_vector_type(16))) __bf16          bf16x16;
typedef __attribute__((ext_vector_type(16))) unsigned short  u16x16;
typedef __attribute__((ext_vector_type(8)))  unsigned short  u16x8;
typedef __attribute__((ext_vector_type(8)))  float           v8f;

__device__ __forceinline__ unsigned short f2bf_bits(float f) {
    __bf16 b = (__bf16)f;                       // hardware f32->bf16 cvt (RNE)
    union { __bf16 b; unsigned short s; } o; o.b = b;
    return o.s;
}

// ---------------------------------------------------------------------------
__global__ void zero_kernel(float* p, unsigned long long n) {
    unsigned long long i = (unsigned long long)blockIdx.x * blockDim.x + threadIdx.x;
    unsigned long long st = (unsigned long long)gridDim.x * blockDim.x;
    for (; i < n; i += st) p[i] = 0.0f;
}

// Area-weighted face normals scattered to vertices.
__global__ void face_normals_kernel(const float* __restrict__ verts,
                                    const int* __restrict__ f0,
                                    const int* __restrict__ f1,
                                    const int* __restrict__ f2,
                                    float* __restrict__ nacc, int F) {
    int f = blockIdx.x * blockDim.x + threadIdx.x;
    if (f >= F) return;
    int i0 = f0[f], i1 = f1[f], i2 = f2[f];
    float ax = verts[3*i0+0], ay = verts[3*i0+1], az = verts[3*i0+2];
    float e1x = verts[3*i1+0]-ax, e1y = verts[3*i1+1]-ay, e1z = verts[3*i1+2]-az;
    float e2x = verts[3*i2+0]-ax, e2y = verts[3*i2+1]-ay, e2z = verts[3*i2+2]-az;
    float cx = e1y*e2z - e1z*e2y;
    float cy = e1z*e2x - e1x*e2z;
    float cz = e1x*e2y - e1y*e2x;
    atomicAdd(&nacc[3*i0+0], cx); atomicAdd(&nacc[3*i0+1], cy); atomicAdd(&nacc[3*i0+2], cz);
    atomicAdd(&nacc[3*i1+0], cx); atomicAdd(&nacc[3*i1+1], cy); atomicAdd(&nacc[3*i1+2], cz);
    atomicAdd(&nacc[3*i2+0], cx); atomicAdd(&nacc[3*i2+1], cy); atomicAdd(&nacc[3*i2+2], cz);
}

__global__ void normalize_kernel(float* nacc, int N) {
    int v = blockIdx.x * blockDim.x + threadIdx.x;
    if (v >= N) return;
    float x = nacc[3*v+0], y = nacc[3*v+1], z = nacc[3*v+2];
    float inv = 1.0f / (sqrtf(x*x + y*y + z*z) + 1e-8f);
    nacc[3*v+0] = x*inv; nacc[3*v+1] = y*inv; nacc[3*v+2] = z*inv;
}

// features [N,CIN] f32 -> bf16 row-major; 8 elements per thread.
__global__ void prep_featbf_kernel(const float* __restrict__ feat,
                                   unsigned short* __restrict__ featbf,
                                   long long n8) {
    long long idx = (long long)blockIdx.x * blockDim.x + threadIdx.x;
    if (idx >= n8) return;
    const float4* fp = (const float4*)(feat + idx * 8);
    float4 a = fp[0], b = fp[1];
    u16x8 o;
    o[0] = f2bf_bits(a.x); o[1] = f2bf_bits(a.y); o[2] = f2bf_bits(a.z); o[3] = f2bf_bits(a.w);
    o[4] = f2bf_bits(b.x); o[5] = f2bf_bits(b.y); o[6] = f2bf_bits(b.z); o[7] = f2bf_bits(b.w);
    *(u16x8*)(featbf + idx * 8) = o;
}

// Wlin [CIN,COUT] f32 -> WlinT [COUT,CIN] bf16 (each WMMA B lane then reads 16
// contiguous K values with one 32-byte load).
__global__ void prep_wlint_kernel(const float* __restrict__ wlin,
                                  unsigned short* __restrict__ wlinT) {
    int idx = blockIdx.x * blockDim.x + threadIdx.x;   // 128*128
    if (idx >= CIN * COUT) return;
    int c = idx >> 7, k = idx & 127;
    wlinT[(size_t)c * CIN + k] = f2bf_bits(wlin[(size_t)k * COUT + c]);
}

// ---------------------------------------------------------------------------
// x = features @ Wlin + blin via v_wmma_f32_16x16x32_bf16.
// One wave per 16-row strip; 8 column tiles (COUT=128); K in 4 steps of 32.
// A and B are pre-converted bf16, so the loop is b128 loads + WMMA only.
__global__ void __launch_bounds__(128)
gemm_wmma_kernel(const unsigned short* __restrict__ featbf,
                 const unsigned short* __restrict__ wlinT,
                 const float* __restrict__ blin,
                 float* __restrict__ x, int N) {
    const int lane = threadIdx.x & 31;
    const int strip = blockIdx.x * 4 + (threadIdx.x >> 5);
    const int rowBase = strip * 16;
    if (rowBase >= N) return;
    const bool up = lane >= 16;
    const int  rl = lane & 15;
    int r = rowBase + rl; if (r >= N) r = N - 1;

    v8f acc[8] = {};
    #pragma unroll
    for (int ks = 0; ks < CIN; ks += 32) {
        // A fragment: lanes 0-15 hold K {ks..ks+15} of row rl,
        // lanes 16-31 hold K {ks+16..ks+31}.  Two 16B loads.
        const unsigned short* ap = featbf + (size_t)r * CIN + ks + (up ? 16 : 0);
        u16x8 a0 = *(const u16x8*)(ap + 0);
        u16x8 a1 = *(const u16x8*)(ap + 8);
        u16x16 ar;
        #pragma unroll
        for (int i = 0; i < 8; ++i) { ar[i] = a0[i]; ar[8 + i] = a1[i]; }
        bf16x16 afrag = __builtin_bit_cast(bf16x16, ar);
        #pragma unroll
        for (int ct = 0; ct < 8; ++ct) {
            // B fragment: lane holds column (ct*16+rl), 16 contiguous K values
            // starting at ks (+16 for the upper lane half).  One 32B load.
            const int col = ct * 16 + rl;
            const unsigned short* bp = wlinT + (size_t)col * CIN + ks + (up ? 16 : 0);
            u16x16 braw = *(const u16x16*)bp;
            bf16x16 bfrag = __builtin_bit_cast(bf16x16, braw);
            acc[ct] = __builtin_amdgcn_wmma_f32_16x16x32_bf16(
                false, afrag, false, bfrag, (short)0, acc[ct], false, false);
        }
    }
    #pragma unroll
    for (int ct = 0; ct < 8; ++ct) {
        const int col = ct * 16 + rl;
        const float bias = blin[col];
        #pragma unroll
        for (int i = 0; i < 8; ++i) {
            int row = rowBase + i + (up ? 8 : 0);
            if (row < N) x[(size_t)row * COUT + col] = acc[ct][i] + bias;
        }
    }
}

// ---------------------------------------------------------------------------
// Per-edge metric weight: t = tangent projection, u = S(theta) t, w = exp(-|u|^2)
__global__ void edge_weight_kernel(const float* __restrict__ verts,
                                   const float* __restrict__ normals,
                                   const int* __restrict__ src,
                                   const int* __restrict__ dst,
                                   const float* __restrict__ W1,
                                   const float* __restrict__ b1,
                                   const float* __restrict__ W2,
                                   const float* __restrict__ b2,
                                   float* __restrict__ w, int E) {
    __shared__ float sW1[3 * HDIM];
    __shared__ float sb1[HDIM];
    __shared__ float sW2[HDIM * 6];
    __shared__ float sb2[6];
    for (int i = threadIdx.x; i < 3 * HDIM;   i += blockDim.x) sW1[i] = W1[i];
    for (int i = threadIdx.x; i < HDIM;       i += blockDim.x) sb1[i] = b1[i];
    for (int i = threadIdx.x; i < HDIM * 6;   i += blockDim.x) sW2[i] = W2[i];
    for (int i = threadIdx.x; i < 6;          i += blockDim.x) sb2[i] = b2[i];
    __syncthreads();

    int e = blockIdx.x * blockDim.x + threadIdx.x;
    if (e >= E) return;
    int s = src[e], d = dst[e];
    float dx = verts[3*d+0]-verts[3*s+0];
    float dy = verts[3*d+1]-verts[3*s+1];
    float dz = verts[3*d+2]-verts[3*s+2];
    float nx = normals[3*s+0], ny = normals[3*s+1], nz = normals[3*s+2];
    float dn = dx*nx + dy*ny + dz*nz;
    float tx = dx - dn*nx, ty = dy - dn*ny, tz = dz - dn*nz;

    float th[6];
    #pragma unroll
    for (int k = 0; k < 6; ++k) th[k] = sb2[k];
    #pragma unroll 4
    for (int j = 0; j < HDIM; ++j) {
        float h = fmaf(tx, sW1[0*HDIM+j], fmaf(ty, sW1[1*HDIM+j], fmaf(tz, sW1[2*HDIM+j], sb1[j])));
        h = fmaxf(h, 0.0f);
        #pragma unroll
        for (int k = 0; k < 6; ++k) th[k] = fmaf(h, sW2[j*6+k], th[k]);
    }
    // S = [[t0,t1,t2],[t1,t3,t4],[t2,t4,t5]];  M = S*S  =>  q = |S t|^2
    float u0 = th[0]*tx + th[1]*ty + th[2]*tz;
    float u1 = th[1]*tx + th[3]*ty + th[4]*tz;
    float u2 = th[2]*tx + th[4]*ty + th[5]*tz;
    w[e] = expf(-(u0*u0 + u1*u1 + u2*u2));
}

// One wave per edge; 4 channels per lane; atomics land in L2 (num fits in 192MB).
__global__ void scatter_kernel(const int* __restrict__ src,
                               const int* __restrict__ dst,
                               const float* __restrict__ w,
                               const float* __restrict__ x,
                               float* __restrict__ num,
                               float* __restrict__ den, int E) {
    int gwave = (blockIdx.x * blockDim.x + threadIdx.x) >> 5;
    int lane  = threadIdx.x & 31;
    if (gwave >= E) return;
    int s = src[gwave], d = dst[gwave];
    float we = w[gwave];
    const float4 xv = *(const float4*)(x + (size_t)s * COUT + lane * 4);
    float* np = num + (size_t)d * COUT + lane * 4;
    atomicAdd(np + 0, we * xv.x);
    atomicAdd(np + 1, we * xv.y);
    atomicAdd(np + 2, we * xv.z);
    atomicAdd(np + 3, we * xv.w);
    if (lane == 0) atomicAdd(&den[d], we);
}

// out0 = num/(den+eps); accumulate per-column sum / sumsq.
__global__ void colstats_kernel(const float* __restrict__ num,
                                const float* __restrict__ den,
                                float* __restrict__ out0,
                                float* __restrict__ sums,
                                float* __restrict__ sumsq, int N) {
    __shared__ float l1[COUT];
    __shared__ float l2[COUT];
    int tid = threadIdx.x;
    int c = tid & 127;
    int base = blockIdx.x * 128;
    float s = 0.0f, s2 = 0.0f;
    for (int r = base + (tid >> 7); r < base + 128 && r < N; r += 2) {
        float o = num[(size_t)r * COUT + c] / (den[r] + 1e-5f);
        out0[(size_t)r * COUT + c] = o;
        s += o; s2 += o * o;
    }
    if (tid >= 128) { l1[c] = s; l2[c] = s2; }
    __syncthreads();
    if (tid < 128) {
        s  += l1[c];
        s2 += l2[c];
        atomicAdd(&sums[c],  s);
        atomicAdd(&sumsq[c], s2);
    }
}

// (out - mean)/(std+eps), ddof=1, then ELU.
__global__ void final_kernel(float* __restrict__ out,
                             const float* __restrict__ sums,
                             const float* __restrict__ sumsq,
                             long long total, int N) {
    long long i = (long long)blockIdx.x * blockDim.x + threadIdx.x;
    if (i >= total) return;
    int c = (int)(i & 127);
    float sum = sums[c], sq = sumsq[c];
    float mean = sum / (float)N;
    float var  = (sq - mean * mean * (float)N) / (float)(N - 1);
    float sd   = sqrtf(fmaxf(var, 0.0f));
    float y = (out[i] - mean) / (sd + 1e-5f);
    out[i] = (y > 0.0f) ? y : (expf(y) - 1.0f);
}

// ---------------------------------------------------------------------------
extern "C" void kernel_launch(void* const* d_in, const int* in_sizes, int n_in,
                              void* d_out, int out_size, void* d_ws, size_t ws_size,
                              hipStream_t stream) {
    const float* features = (const float*)d_in[0];
    const float* vertices = (const float*)d_in[1];
    const int*   edges    = (const int*)d_in[2];
    const int*   faces    = (const int*)d_in[3];
    const float* W1   = (const float*)d_in[4];
    const float* b1   = (const float*)d_in[5];
    const float* W2   = (const float*)d_in[6];
    const float* b2   = (const float*)d_in[7];
    const float* Wlin = (const float*)d_in[8];
    const float* blin = (const float*)d_in[9];
    float* out = (float*)d_out;

    const int N = in_sizes[0] / CIN;      // 100000
    const int E = in_sizes[2] / 2;        // 600000
    const int F = in_sizes[3] / 3;        // 200000

    const int* e_src = edges;
    const int* e_dst = edges + E;
    const int* f0 = faces;
    const int* f1 = faces + F;
    const int* f2 = faces + 2 * F;

    // Workspace layout (float units). Front section is zero-initialized each call.
    float* wsf = (float*)d_ws;
    size_t offNorm = 0;                                   // N*3
    size_t offNum  = offNorm + (size_t)N * 3;             // N*COUT
    size_t offDen  = offNum  + (size_t)N * COUT;          // N
    size_t offSums = offDen  + (size_t)N;                 // 128
    size_t offSq   = offSums + 128;                       // 128
    size_t zeroCnt = offSq   + 128;
    size_t offW    = zeroCnt;                             // E
    size_t offX    = (offW + (size_t)E + 7) & ~(size_t)7;            // N*COUT f32
    size_t offWT   = (offX + (size_t)N * COUT + 7) & ~(size_t)7;     // 128*128 bf16
    size_t offFB   = (offWT + (size_t)CIN * COUT / 2 + 7) & ~(size_t)7; // N*CIN bf16

    float* nacc  = wsf + offNorm;
    float* num   = wsf + offNum;
    float* den   = wsf + offDen;
    float* sums  = wsf + offSums;
    float* sumsq = wsf + offSq;
    float* wbuf  = wsf + offW;
    float* xbuf  = wsf + offX;
    unsigned short* wlinT  = (unsigned short*)(wsf + offWT);
    unsigned short* featbf = (unsigned short*)(wsf + offFB);

    // 1. zero accumulators
    zero_kernel<<<4096, 256, 0, stream>>>(wsf, (unsigned long long)zeroCnt);
    // 2. face normals -> vertex accumulation
    face_normals_kernel<<<(F + 255) / 256, 256, 0, stream>>>(vertices, f0, f1, f2, nacc, F);
    // 3. normalize vertex normals (in place)
    normalize_kernel<<<(N + 255) / 256, 256, 0, stream>>>(nacc, N);
    // 4. features -> bf16, Wlin -> bf16 transpose
    {
        long long n8 = (long long)N * CIN / 8;
        prep_featbf_kernel<<<(int)((n8 + 255) / 256), 256, 0, stream>>>(features, featbf, n8);
    }
    prep_wlint_kernel<<<(CIN * COUT + 255) / 256, 256, 0, stream>>>(Wlin, wlinT);
    // 5. feature GEMM on WMMA (bf16 in, f32 acc)
    {
        int strips = (N + 15) / 16;
        int blocks = (strips + 3) / 4;
        gemm_wmma_kernel<<<blocks, 128, 0, stream>>>(featbf, wlinT, blin, xbuf, N);
    }
    // 6. per-edge metric weights
    edge_weight_kernel<<<(E + 255) / 256, 256, 0, stream>>>(
        vertices, nacc, e_src, e_dst, W1, b1, W2, b2, wbuf, E);
    // 7. weighted scatter (1 wave per edge)
    {
        long long threads = (long long)E * 32;
        int blocks = (int)((threads + 255) / 256);
        scatter_kernel<<<blocks, 256, 0, stream>>>(e_src, e_dst, wbuf, xbuf, num, den, E);
    }
    // 8. out0 = num/(den+eps) + column stats
    colstats_kernel<<<(N + 127) / 128, 256, 0, stream>>>(num, den, out, sums, sumsq, N);
    // 9. normalize + ELU (in place on d_out)
    {
        long long total = (long long)N * COUT;
        int blocks = (int)((total + 255) / 256);
        final_kernel<<<blocks, 256, 0, stream>>>(out, sums, sumsq, total, N);
    }
}
